// Decoder_Processor_79731772883064
// MI455X (gfx1250) — matrix operand order repaired and checked by reading, not compiled, not run on hardware
//
#include <hip/hip_runtime.h>
#include <math.h>

// ---------------- problem constants ----------------
#define Hn   512
#define Dn   8
#define Bn   64
#define Tn   256
#define An   128
#define EPSf 1e-5f

// ---------------- WMMA types ----------------
typedef __bf16 bf16x16 __attribute__((ext_vector_type(16)));
typedef float  f32x8   __attribute__((ext_vector_type(8)));

__device__ __forceinline__ unsigned short f2bf(float f) {
    unsigned u = __float_as_uint(f);
    u += 0x7FFFu + ((u >> 16) & 1u);        // round-to-nearest-even
    return (unsigned short)(u >> 16);
}

// A-fragment: 16x32 bf16, activations row-major [rows][ld] (global or LDS).
// lane l<16 : row m0+l, halves[0..15]=K k0..k0+15
// lane 16+l: row m0+l, halves[0..15]=K k0+16..k0+31
__device__ __forceinline__ bf16x16 load_a_frag(const unsigned short* base, int m0,
                                               int ld, int k0, int lane) {
    int lo = lane & 15, hi = lane >> 4;
    const unsigned short* p = base + (size_t)(m0 + lo) * ld + k0 + hi * 16;
    union { bf16x16 v; uint4 u[2]; } t;
    t.u[0] = *(const uint4*)(p);
    t.u[1] = *(const uint4*)(p + 8);
    return t.v;
}

// B-fragment: 32x16 bf16 pre-tiled: tile = 512 bf16, lane gets 16 contiguous.
__device__ __forceinline__ bf16x16 load_b_frag(const unsigned short* tiled,
                                               int tileIdx, int lane) {
    const unsigned short* p = tiled + (size_t)tileIdx * 512 + lane * 16;
    union { bf16x16 v; uint4 u[2]; } t;
    t.u[0] = *(const uint4*)(p);
    t.u[1] = *(const uint4*)(p + 8);
    return t.v;
}

__device__ __forceinline__ f32x8 wmma_bf16(bf16x16 a, bf16x16 b, f32x8 c) {
    return __builtin_amdgcn_wmma_f32_16x16x32_bf16(false, a, false, b,
                                                   (short)0, c, false, false);
}

// ---- async stage: global -> LDS via CDNA5 async copy (ASYNCcnt tracked) ----
__device__ __forceinline__ void stage_async(const unsigned short* gsrc,
                                            unsigned short* sdst, int bytes) {
    unsigned ldsBase = (unsigned)(uintptr_t)sdst;      // low 32 bits = LDS addr
    const char* g = (const char*)gsrc;
    for (int c = threadIdx.x * 16; c < bytes; c += blockDim.x * 16) {
        unsigned lds = ldsBase + c;
        unsigned long long ga = (unsigned long long)(uintptr_t)(g + c);
        asm volatile("global_load_async_to_lds_b128 %0, %1, off"
                     :: "v"(lds), "v"(ga) : "memory");
    }
}
__device__ __forceinline__ void stage_wait() {
    asm volatile("s_wait_asynccnt 0x0" ::: "memory");
    __syncthreads();
}

__device__ __forceinline__ float block_reduce_sum(float v, float* red) {
    int tid = threadIdx.x;
    red[tid] = v; __syncthreads();
    for (int s = 128; s > 0; s >>= 1) {
        if (tid < s) red[tid] += red[tid + s];
        __syncthreads();
    }
    float r = red[0]; __syncthreads();
    return r;
}

// ---------------- one-time: tile f32 weights [K][N] -> bf16 WMMA B-tiles ----
__global__ void tile_weights_kernel(const float* __restrict__ src,
                                    unsigned short* __restrict__ dst,
                                    int K, int N) {
    int e = blockIdx.x * blockDim.x + threadIdx.x;
    if (e >= K * N) return;
    int tile = e >> 9, within = e & 511;
    int lane = within >> 4, hj = within & 15;
    int NT = N >> 4;
    int kt = tile / NT, nt = tile % NT;
    int k = kt * 32 + ((lane >> 4) << 4) + hj;
    int n = nt * 16 + (lane & 15);
    dst[e] = f2bf(src[(size_t)k * N + n]);
}

// ---------------- per step: x[:,t,:] -> bf16 (also layer-0 carry) ----------
__global__ void xstep_kernel(const float* __restrict__ x, int t,
                             unsigned short* __restrict__ xcBF) {
    int idx = blockIdx.x * blockDim.x + threadIdx.x;
    if (idx >= Bn * Hn) return;
    int b = idx / Hn, j = idx % Hn;
    xcBF[idx] = f2bf(x[((size_t)b * Tn + t) * Hn + j]);
}

// ---------------- action GEMMs: all 8 scan steps in parallel ----------------
// block = scan index i; 8 waves each own one N-tile; 4 M-tiles per wave.
// P[i] = relu(prev_i @ W1 + cur_i @ U1 + b1);  prev_7 = x_step
__global__ void action_gemm_kernel(const unsigned short* __restrict__ hbOld,
                                   const unsigned short* __restrict__ xstepBF,
                                   const unsigned short* __restrict__ W1t,
                                   const unsigned short* __restrict__ U1t,
                                   const float* __restrict__ ba1,
                                   float* __restrict__ P) {
    extern __shared__ unsigned short smem[];           // 2 x 64x512 bf16
    unsigned short* sPrev = smem;
    unsigned short* sCur  = smem + Bn * Hn;
    int i    = blockIdx.x;
    int lane = threadIdx.x & 31;
    int nt   = threadIdx.x >> 5;                       // 0..7

    const unsigned short* prev = (i < 7) ? hbOld + (size_t)(6 - i) * Bn * Hn : xstepBF;
    const unsigned short* cur  = hbOld + (size_t)(7 - i) * Bn * Hn;
    stage_async(prev, sPrev, Bn * Hn * 2);
    stage_async(cur,  sCur,  Bn * Hn * 2);
    stage_wait();

    f32x8 acc0 = {}, acc1 = {}, acc2 = {}, acc3 = {};
    for (int kt = 0; kt < 16; ++kt) {
        if (kt + 1 < 16) {
            __builtin_prefetch(W1t + (size_t)((kt + 1) * 8 + nt) * 512, 0, 3);
            __builtin_prefetch(U1t + (size_t)((kt + 1) * 8 + nt) * 512, 0, 3);
        }
        bf16x16 bw = load_b_frag(W1t, kt * 8 + nt, lane);
        acc0 = wmma_bf16(load_a_frag(sPrev,  0, Hn, kt * 32, lane), bw, acc0);
        acc1 = wmma_bf16(load_a_frag(sPrev, 16, Hn, kt * 32, lane), bw, acc1);
        acc2 = wmma_bf16(load_a_frag(sPrev, 32, Hn, kt * 32, lane), bw, acc2);
        acc3 = wmma_bf16(load_a_frag(sPrev, 48, Hn, kt * 32, lane), bw, acc3);
        bf16x16 bu = load_b_frag(U1t, kt * 8 + nt, lane);
        acc0 = wmma_bf16(load_a_frag(sCur,  0, Hn, kt * 32, lane), bu, acc0);
        acc1 = wmma_bf16(load_a_frag(sCur, 16, Hn, kt * 32, lane), bu, acc1);
        acc2 = wmma_bf16(load_a_frag(sCur, 32, Hn, kt * 32, lane), bu, acc2);
        acc3 = wmma_bf16(load_a_frag(sCur, 48, Hn, kt * 32, lane), bu, acc3);
    }
    int lo = lane & 15, hi = lane >> 4;
    int col = nt * 16 + lo;
    float bias = ba1[col];
    f32x8 accs[4] = {acc0, acc1, acc2, acc3};
    for (int mt = 0; mt < 4; ++mt)
        for (int r = 0; r < 8; ++r) {
            int row = mt * 16 + hi * 8 + r;
            float v = accs[mt][r] + bias;
            P[((size_t)i * Bn + row) * An + col] = v > 0.0f ? v : 0.0f;
        }
}

// ------ decisions + prefix-AND + masked bf16 h_d for all 8 layers ----------
__global__ void decide_kernel(const float* __restrict__ P,
                              const float* __restrict__ W2,
                              const float* __restrict__ b2,
                              const float* __restrict__ hOldF,
                              int* __restrict__ act, int* __restrict__ proc,
                              unsigned short* __restrict__ hdBF) {
    int rb = blockIdx.x;                   // batch row 0..63
    int tid = threadIdx.x;
    __shared__ int dec[8];
    __shared__ int aSh[8];
    __shared__ int pSh[8];
    int i = tid >> 5, lane = tid & 31;
    float d0 = 0.f, d1 = 0.f;
    for (int c = lane; c < An; c += 32) {
        float p = P[((size_t)i * Bn + rb) * An + c];
        d0 += p * W2[c * 2 + 0];
        d1 += p * W2[c * 2 + 1];
    }
    for (int m = 16; m > 0; m >>= 1) {
        d0 += __shfl_xor(d0, m, 32);
        d1 += __shfl_xor(d1, m, 32);
    }
    if (lane == 0) {
        float e0 = fminf(expf(d0 + b2[0]), 1000.0f);
        float e1 = fminf(expf(d1 + b2[1]), 1000.0f);
        dec[i] = (e0 <= e1) ? 1 : 0;
    }
    __syncthreads();
    if (tid == 0) {
        int a = 1;
        for (int k = 0; k < 8; ++k) { a &= dec[k]; aSh[k] = a; }
        for (int k = 0; k < 7; ++k) pSh[k] = aSh[k + 1];
        pSh[7] = 1;
        for (int k = 0; k < 8; ++k) {
            act[k * Bn + rb]  = aSh[k];
            proc[k * Bn + rb] = pSh[k];
        }
    }
    __syncthreads();
    for (int e = tid; e < Dn * Hn; e += blockDim.x) {
        int d = e / Hn, j = e % Hn;
        float h = aSh[d] ? 0.0f : hOldF[((size_t)d * Bn + rb) * Hn + j];
        hdBF[((size_t)d * Bn + rb) * Hn + j] = f2bf(h);
    }
}

// ---------------- layer phase 1: pre1 = x_c@W + b ; pre2 = h_d@U[:,:1024] ---
// blocks 0..11 -> W part (96 N-tiles), blocks 12..19 -> U part (64 N-tiles).
// Each wave owns one N-tile, computes all 4 M-tiles from LDS-staged A panel.
__global__ void phase1_gemm_kernel(const unsigned short* __restrict__ xcBF,
                                   const unsigned short* __restrict__ hdBF, int d,
                                   const unsigned short* __restrict__ Wt,
                                   const unsigned short* __restrict__ Ut,
                                   const float* __restrict__ bias,
                                   float* __restrict__ pre1,
                                   float* __restrict__ pre2) {
    extern __shared__ unsigned short sA[];             // 64x512 bf16 panel
    int lane = threadIdx.x & 31;
    int wv   = threadIdx.x >> 5;                       // 0..7
    bool isW = blockIdx.x < 12;
    int nt   = isW ? blockIdx.x * 8 + wv : (blockIdx.x - 12) * 8 + wv;
    const unsigned short* Am = isW ? xcBF : hdBF + (size_t)d * Bn * Hn;
    const unsigned short* Bt = isW ? Wt : Ut;
    float* outp = isW ? pre1 : pre2;
    int ldout   = isW ? 3 * Hn : 2 * Hn;

    stage_async(Am, sA, Bn * Hn * 2);
    stage_wait();

    f32x8 acc0 = {}, acc1 = {}, acc2 = {}, acc3 = {};
    for (int kt = 0; kt < 16; ++kt) {
        if (kt + 1 < 16)
            __builtin_prefetch(Bt + (size_t)((kt + 1) * 96 + nt) * 512, 0, 3);
        bf16x16 bm = load_b_frag(Bt, kt * 96 + nt, lane);
        acc0 = wmma_bf16(load_a_frag(sA,  0, Hn, kt * 32, lane), bm, acc0);
        acc1 = wmma_bf16(load_a_frag(sA, 16, Hn, kt * 32, lane), bm, acc1);
        acc2 = wmma_bf16(load_a_frag(sA, 32, Hn, kt * 32, lane), bm, acc2);
        acc3 = wmma_bf16(load_a_frag(sA, 48, Hn, kt * 32, lane), bm, acc3);
    }
    int lo = lane & 15, hi = lane >> 4;
    int col = nt * 16 + lo;
    float bv = isW ? bias[col] : 0.0f;
    f32x8 accs[4] = {acc0, acc1, acc2, acc3};
    for (int mt = 0; mt < 4; ++mt)
        for (int r = 0; r < 8; ++r) {
            int row = mt * 16 + hi * 8 + r;
            outp[(size_t)row * ldout + col] = accs[mt][r] + bv;
        }
}

// ------- phase 1b: LN(pre1), LN(pre2), gates z,r; rh = r*h_d (bf16) --------
__global__ void phase1_ln_kernel(const float* __restrict__ pre1,
                                 const float* __restrict__ pre2,
                                 const float* __restrict__ gam,
                                 const float* __restrict__ bet,
                                 const float* __restrict__ hOldF,
                                 const int* __restrict__ act, int d,
                                 float* __restrict__ zbuf,
                                 unsigned short* __restrict__ rhBF,
                                 float* __restrict__ s1c) {
    __shared__ float red[256];
    int row = blockIdx.x, tid = threadIdx.x;
    const float* p1 = pre1 + (size_t)row * (3 * Hn);
    const float* p2 = pre2 + (size_t)row * (2 * Hn);
    float s1 = 0.f, q1 = 0.f, s2 = 0.f, q2 = 0.f;
    for (int j = tid; j < 3 * Hn; j += 256) { float v = p1[j]; s1 += v; q1 += v * v; }
    for (int j = tid; j < 2 * Hn; j += 256) { float v = p2[j]; s2 += v; q2 += v * v; }
    float S1 = block_reduce_sum(s1, red);
    float Q1 = block_reduce_sum(q1, red);
    float S2 = block_reduce_sum(s2, red);
    float Q2 = block_reduce_sum(q2, red);
    float m1 = S1 / (3 * Hn), var1 = Q1 / (3 * Hn) - m1 * m1;
    float m2 = S2 / (2 * Hn), var2 = Q2 / (2 * Hn) - m2 * m2;
    float inv1 = 1.0f / (sqrtf(var1 + EPSf) + EPSf);
    float inv2 = 1.0f / (sqrtf(var2 + EPSf) + EPSf);
    const float* g0 = gam;            const float* b0 = bet;
    const float* g1 = gam + 3 * Hn;   const float* b1 = bet + 3 * Hn;
    int aflag = act[d * Bn + row];
    for (int j = tid; j < 3 * Hn; j += 256) {
        float ln1 = g0[j] * ((p1[j] - m1) * inv1) + b0[j];
        if (j < 2 * Hn) {
            float ln2 = g1[j] * ((p2[j] - m2) * inv2) + b1[j];
            float s = 0.2f * (ln1 + ln2) + 0.5f;
            s = fminf(fmaxf(s, 0.0f), 1.0f);
            if (j < Hn) {
                zbuf[(size_t)row * Hn + j] = s;
            } else {
                int jj = j - Hn;
                float hd = aflag ? 0.0f : hOldF[((size_t)d * Bn + row) * Hn + jj];
                rhBF[(size_t)row * Hn + jj] = f2bf(s * hd);
            }
        } else {
            s1c[(size_t)row * Hn + (j - 2 * Hn)] = ln1;
        }
    }
}

// ---------------- layer phase 2: pre3 = (r*h_d) @ U[:,1024:1536] -----------
__global__ void phase2_gemm_kernel(const unsigned short* __restrict__ rhBF,
                                   const unsigned short* __restrict__ Ut,
                                   float* __restrict__ pre3) {
    extern __shared__ unsigned short sA[];             // 64x512 bf16 panel
    int lane = threadIdx.x & 31;
    int wv   = threadIdx.x >> 5;
    int nt   = blockIdx.x * 8 + wv;                    // 0..31
    int ntg  = 64 + nt;                                // N columns 1024..1535

    stage_async(rhBF, sA, Bn * Hn * 2);
    stage_wait();

    f32x8 acc0 = {}, acc1 = {}, acc2 = {}, acc3 = {};
    for (int kt = 0; kt < 16; ++kt) {
        if (kt + 1 < 16)
            __builtin_prefetch(Ut + (size_t)((kt + 1) * 96 + ntg) * 512, 0, 3);
        bf16x16 bm = load_b_frag(Ut, kt * 96 + ntg, lane);
        acc0 = wmma_bf16(load_a_frag(sA,  0, Hn, kt * 32, lane), bm, acc0);
        acc1 = wmma_bf16(load_a_frag(sA, 16, Hn, kt * 32, lane), bm, acc1);
        acc2 = wmma_bf16(load_a_frag(sA, 32, Hn, kt * 32, lane), bm, acc2);
        acc3 = wmma_bf16(load_a_frag(sA, 48, Hn, kt * 32, lane), bm, acc3);
    }
    int lo = lane & 15, hi = lane >> 4;
    int col = nt * 16 + lo;
    f32x8 accs[4] = {acc0, acc1, acc2, acc3};
    for (int mt = 0; mt < 4; ++mt)
        for (int r = 0; r < 8; ++r) {
            int row = mt * 16 + hi * 8 + r;
            pre3[(size_t)row * Hn + col] = accs[mt][r];
        }
}

// ------- phase 2b: LN(pre3) -> cand -> combine -> new h, next carry --------
__global__ void phase2_ln_kernel(const float* __restrict__ pre3,
                                 const float* __restrict__ gam,
                                 const float* __restrict__ bet,
                                 const float* __restrict__ s1c,
                                 const float* __restrict__ zbuf,
                                 const float* __restrict__ hOldF,
                                 const int* __restrict__ act,
                                 const int* __restrict__ proc, int d,
                                 float* __restrict__ hNewF,
                                 unsigned short* __restrict__ hNewBF,
                                 unsigned short* __restrict__ xcBF,
                                 float* __restrict__ out, int t) {
    __shared__ float red[256];
    int row = blockIdx.x, tid = threadIdx.x;
    const float* p3 = pre3 + (size_t)row * Hn;
    float s = 0.f, q = 0.f;
    for (int j = tid; j < Hn; j += 256) { float v = p3[j]; s += v; q += v * v; }
    float S = block_reduce_sum(s, red);
    float Q = block_reduce_sum(q, red);
    float m = S / Hn, var = Q / Hn - m * m;
    float inv = 1.0f / (sqrtf(var + EPSf) + EPSf);
    const float* g1 = gam + 3 * Hn;
    const float* b1 = bet + 3 * Hn;
    int aflag = act[d * Bn + row];
    int pflag = proc[d * Bn + row];
    for (int j = tid; j < Hn; j += 256) {
        float ln3 = g1[2 * Hn + j] * ((p3[j] - m) * inv) + b1[2 * Hn + j];
        float cand = tanhf(s1c[(size_t)row * Hn + j] + ln3);
        float hd = aflag ? 0.0f : hOldF[((size_t)d * Bn + row) * Hn + j];
        float z  = zbuf[(size_t)row * Hn + j];
        float hnew = z * hd + (1.0f - z) * cand;
        float hout = pflag ? hnew : hd;
        hNewF [((size_t)d * Bn + row) * Hn + j] = hout;
        hNewBF[((size_t)d * Bn + row) * Hn + j] = f2bf(hout);
        xcBF[(size_t)row * Hn + j] = f2bf(hout);
        if (d == Dn - 1)
            out[((size_t)row * Tn + t) * Hn + j] = hout;
    }
}

// =====================================================================
extern "C" void kernel_launch(void* const* d_in, const int* in_sizes, int n_in,
                              void* d_out, int out_size, void* d_ws, size_t ws_size,
                              hipStream_t stream) {
    (void)in_sizes; (void)n_in; (void)out_size; (void)ws_size;
    const float* x   = (const float*)d_in[0];
    const float* W   = (const float*)d_in[1];
    const float* U   = (const float*)d_in[2];
    const float* bb  = (const float*)d_in[3];
    const float* W1  = (const float*)d_in[4];
    const float* U1  = (const float*)d_in[5];
    const float* ba1 = (const float*)d_in[6];
    const float* W2  = (const float*)d_in[7];
    const float* b2  = (const float*)d_in[8];
    const float* gam = (const float*)d_in[9];
    const float* bet = (const float*)d_in[10];
    float* out = (float*)d_out;

    // ---- workspace layout ----
    char* ws = (char*)d_ws;
    size_t off = 0;
    auto alloc = [&](size_t bytes) { void* p = ws + off; off += (bytes + 255) & ~size_t(255); return p; };
    unsigned short* Wt   = (unsigned short*)alloc((size_t)Hn * 3 * Hn * 2);
    unsigned short* Ut   = (unsigned short*)alloc((size_t)Hn * 3 * Hn * 2);
    unsigned short* W1t  = (unsigned short*)alloc((size_t)Hn * An * 2);
    unsigned short* U1t  = (unsigned short*)alloc((size_t)Hn * An * 2);
    float*          hF0  = (float*)alloc((size_t)Dn * Bn * Hn * 4);
    float*          hF1  = (float*)alloc((size_t)Dn * Bn * Hn * 4);
    unsigned short* hB0  = (unsigned short*)alloc((size_t)Dn * Bn * Hn * 2);
    unsigned short* hB1  = (unsigned short*)alloc((size_t)Dn * Bn * Hn * 2);
    unsigned short* hdBF = (unsigned short*)alloc((size_t)Dn * Bn * Hn * 2);
    unsigned short* xcBF = (unsigned short*)alloc((size_t)Bn * Hn * 2);
    float*          P    = (float*)alloc((size_t)Dn * Bn * An * 4);
    int*            act  = (int*)alloc((size_t)Dn * Bn * 4);
    int*            proc = (int*)alloc((size_t)Dn * Bn * 4);
    float*          pre1 = (float*)alloc((size_t)Bn * 3 * Hn * 4);
    float*          pre2 = (float*)alloc((size_t)Bn * 2 * Hn * 4);
    float*          pre3 = (float*)alloc((size_t)Bn * Hn * 4);
    float*          zbuf = (float*)alloc((size_t)Bn * Hn * 4);
    float*          s1c  = (float*)alloc((size_t)Bn * Hn * 4);
    unsigned short* rhBF = (unsigned short*)alloc((size_t)Bn * Hn * 2);

    // h0 = zeros (f32 + bf16 views of buffer 0)
    hipMemsetAsync(hF0, 0, (size_t)Dn * Bn * Hn * 4, stream);
    hipMemsetAsync(hB0, 0, (size_t)Dn * Bn * Hn * 2, stream);

    // one-time weight tiling (pure function of inputs; re-done every call)
    {
        int n;
        n = Hn * 3 * Hn;
        tile_weights_kernel<<<(n + 255) / 256, 256, 0, stream>>>(W, Wt, Hn, 3 * Hn);
        tile_weights_kernel<<<(n + 255) / 256, 256, 0, stream>>>(U, Ut, Hn, 3 * Hn);
        n = Hn * An;
        tile_weights_kernel<<<(n + 255) / 256, 256, 0, stream>>>(W1, W1t, Hn, An);
        tile_weights_kernel<<<(n + 255) / 256, 256, 0, stream>>>(U1, U1t, Hn, An);
    }

    const size_t panelBytes  = (size_t)Bn * Hn * 2;    // 64 KB
    const size_t panel2Bytes = 2 * panelBytes;         // 128 KB

    for (int t = 0; t < Tn; ++t) {
        float*          hOldF = (t & 1) ? hF1 : hF0;
        float*          hNewF = (t & 1) ? hF0 : hF1;
        unsigned short* hOldB = (t & 1) ? hB1 : hB0;
        unsigned short* hNewB = (t & 1) ? hB0 : hB1;

        xstep_kernel<<<(Bn * Hn + 255) / 256, 256, 0, stream>>>(x, t, xcBF);
        action_gemm_kernel<<<8, 256, panel2Bytes, stream>>>(hOldB, xcBF, W1t, U1t,
                                                            ba1, P);
        decide_kernel<<<Bn, 256, 0, stream>>>(P, W2, b2, hOldF, act, proc, hdBF);

        for (int d = 0; d < Dn; ++d) {
            phase1_gemm_kernel<<<20, 256, panelBytes, stream>>>(xcBF, hdBF, d, Wt,
                                                                Ut, bb, pre1, pre2);
            phase1_ln_kernel<<<Bn, 256, 0, stream>>>(pre1, pre2, gam, bet, hOldF,
                                                     act, d, zbuf, rhBF, s1c);
            phase2_gemm_kernel<<<4, 256, panelBytes, stream>>>(rhBF, Ut, pre3);
            phase2_ln_kernel<<<Bn, 256, 0, stream>>>(pre3, gam, bet, s1c, zbuf,
                                                     hOldF, act, proc, d,
                                                     hNewF, hNewB, xcBF, out, t);
        }
    }
}
